// SpMiddleResNetFHD_3178275799071
// MI455X (gfx1250) — compile-verified, matrix operand order checked by reading, not run on hardware
//
#include <hip/hip_runtime.h>
#include <hip/hip_bf16.h>
#include <stdint.h>

typedef unsigned short u16;
typedef __attribute__((ext_vector_type(16))) __bf16 bf16x16;
typedef __attribute__((ext_vector_type(8)))  float   v8f;
typedef __attribute__((ext_vector_type(4)))  int     i32x4;

// address-space qualified pointers for the async global->LDS builtin
typedef __attribute__((address_space(1))) i32x4* as1_i32x4p;
typedef __attribute__((address_space(3))) i32x4* as3_i32x4p;

union FragAB { bf16x16 v; uint4 q[2]; };

#if defined(__HIP_DEVICE_COMPILE__) && __has_builtin(__builtin_amdgcn_global_load_async_to_lds_b128)
#define HAS_ASYNC_LDS 1
#else
#define HAS_ASYNC_LDS 0
#endif

__device__ __forceinline__ float bf2f(u16 h){
  unsigned u = ((unsigned)h) << 16; float f; __builtin_memcpy(&f, &u, 4); return f;
}
__device__ __forceinline__ u16 f2bf(float f){
  unsigned u; __builtin_memcpy(&u, &f, 4);
  u += 0x7FFFu + ((u >> 16) & 1u);            // round-to-nearest-even
  return (u16)(u >> 16);
}

// 16-byte global -> LDS copy: async path when available, sync fallback.
__device__ __forceinline__ void copy16_g2l(const u16* g, u16* l){
#if HAS_ASYNC_LDS
  __builtin_amdgcn_global_load_async_to_lds_b128(
      (as1_i32x4p)(unsigned long long)g,
      (as3_i32x4p)(unsigned)(unsigned long long)l,
      0, 0);
#else
  *(uint4*)l = *(const uint4*)g;
#endif
}
__device__ __forceinline__ void wait_async_lds(){
#if HAS_ASYNC_LDS
#if __has_builtin(__builtin_amdgcn_s_wait_asynccnt)
  __builtin_amdgcn_s_wait_asynccnt(0);
#else
  asm volatile("s_wait_asynccnt 0" ::: "memory");
#endif
#endif
}

// ---------------- utility kernels ----------------
__global__ void zero_f32(float* p, size_t n){
  size_t i = (size_t)blockIdx.x * blockDim.x + threadIdx.x;
  size_t st = (size_t)gridDim.x * blockDim.x;
  for (; i < n; i += st) p[i] = 0.f;
}

__global__ void scatter_k(const float* __restrict__ vf, const int* __restrict__ coors,
                          float* acc, float* cnt, int N){
  int i = blockIdx.x * blockDim.x + threadIdx.x;
  if (i >= N * 16) return;
  int vi = i >> 4, c = i & 15;
  int z = coors[vi*4+1], y = coors[vi*4+2], x = coors[vi*4+3];
  size_t s = ((size_t)z * 200 + y) * 200 + x;
  atomicAdd(&acc[s*16 + c], vf[i]);
  if (c == 0) atomicAdd(&cnt[s], 1.0f);
}

__global__ void finalize_k(const float* __restrict__ acc, const float* __restrict__ cnt,
                           u16* xo, u16* mo, size_t nvox){
  size_t i = (size_t)blockIdx.x * blockDim.x + threadIdx.x;
  if (i >= nvox * 16) return;
  xo[i] = f2bf(acc[i]);
  if ((i & 15) == 0){ size_t v = i >> 4; mo[v] = (cnt[v] > 0.f) ? (u16)0x3F80 : (u16)0; }
}

// OIDHW fp32 -> [tap][cout][CINP] bf16 (zero-padded K)
__global__ void prep_w_k(const float* __restrict__ w, u16* wp,
                         int CIN, int CINP, int COUT, int KD, int KH, int KW){
  int idx = blockIdx.x * blockDim.x + threadIdx.x;
  int total = KD*KH*KW*COUT*CINP;
  if (idx >= total) return;
  int ci = idx % CINP; int t = idx / CINP;
  int co = t % COUT;   int tap = t / COUT;
  int dx = tap % KW;   int t2 = tap / KW; int dy = t2 % KH; int dz = t2 / KH;
  float v = 0.f;
  if (ci < CIN) v = w[(((size_t)co*CIN + ci)*KD + dz)*(KH*KW) + dy*KW + dx];
  wp[idx] = f2bf(v);
}

__global__ void down_mask_k(const u16* __restrict__ mi, u16* mo,
                            int Din,int Hin,int Win,int Dout,int Hout,int Wout,
                            int kd,int kh,int kw,int sd,int sh,int sw,int pd,int ph,int pw){
  int idx = blockIdx.x*blockDim.x + threadIdx.x;
  int total = Dout*Hout*Wout;
  if (idx >= total) return;
  int ox = idx % Wout; int t = idx / Wout; int oy = t % Hout; int oz = t / Hout;
  u16 any = 0;
  for (int dz=0; dz<kd; ++dz){ int iz = oz*sd+dz-pd; if ((unsigned)iz >= (unsigned)Din) continue;
    for (int dy=0; dy<kh; ++dy){ int iy = oy*sh+dy-ph; if ((unsigned)iy >= (unsigned)Hin) continue;
      for (int dx=0; dx<kw; ++dx){ int ix = ox*sw+dx-pw; if ((unsigned)ix >= (unsigned)Win) continue;
        if (mi[((size_t)iz*Hin + iy)*Win + ix]) any = 0x3F80;
  }}}
  mo[idx] = any;
}

// (2,25,25,128) bf16 NDHWC -> (256,25,25) fp32, channel = c*2 + d
__global__ void final_out_k(const u16* __restrict__ xin, float* out){
  int i = blockIdx.x*blockDim.x + threadIdx.x;
  if (i >= 256*25*25) return;
  int w = i % 25; int t = i / 25; int h = t % 25; int cd = t / 25;
  int c = cd >> 1, d = cd & 1;
  out[i] = bf2f(xin[(((size_t)d*25 + h)*25 + w)*128 + c]);
}

// ---------------- fused implicit-GEMM WMMA conv ----------------
// M-tile = 32 output x positions per wave (2 accumulators share each B fragment).
// Double-buffered LDS rows with async global->LDS staging pipelined vs WMMA.
template<int CIN,int COUT,int KD,int KH,int KW,int SD,int SH,int SW>
__global__ void __launch_bounds__(COUT*2)
conv_wmma_k(const u16* __restrict__ in, const u16* __restrict__ wp, u16* __restrict__ out,
            const u16* __restrict__ mask, const u16* __restrict__ resid,
            const float* __restrict__ bias,
            const float* __restrict__ bng, const float* __restrict__ bnb,
            const float* __restrict__ bnm, const float* __restrict__ bnv,
            int Din,int Hin,int Win,int Dout,int Hout,int Wout,int pd,int ph,int pw){
  constexpr int CINP = (CIN < 32) ? 32 : CIN;   // K padded to WMMA K=32
  constexpr int KCH  = CINP / 32;
  constexpr int TW   = 31*SW + KW;              // staged row width (M=32)
  constexpr int NT   = COUT*2;                  // 32 threads per 16 couts
  constexpr int ROWS = KD*KH;
  constexpr int CHUNKS = (TW*CINP)/8;
  __shared__ u16 srow[2][TW * CINP];

  const int x0   = blockIdx.x * 32;
  const int oy   = blockIdx.y;
  const int oz   = blockIdx.z;
  const int tid  = threadIdx.x;
  const int lane = tid & 31;
  const int cout0 = (tid >> 5) * 16;
  const int mrow = lane & 15;                   // M (for A) / N (for B) row
  const int kb   = (lane >> 4) * 8;             // K half select per ISA layout

  auto stage = [&](int buf, int r){
    const int dz = r / KH, dy = r % KH;
    const int iz = oz*SD + dz - pd;
    const int iy = oy*SH + dy - ph;
    const bool rowok = ((unsigned)iz < (unsigned)Din) && ((unsigned)iy < (unsigned)Hin);
    for (int c = tid; c < CHUNKS; c += NT){
      const int p  = c / (CINP/8);
      const int cc = (c % (CINP/8)) * 8;
      const int ix = x0*SW - pw + p;
      u16* lp = &srow[buf][p*CINP + cc];
      if (rowok && (unsigned)ix < (unsigned)Win && cc < CIN){
        copy16_g2l(in + (((size_t)iz*Hin + iy)*Win + ix)*CIN + cc, lp);
      } else {
        *(uint4*)lp = make_uint4(0u,0u,0u,0u);
      }
    }
  };

  v8f acc[2]; acc[0] = {}; acc[1] = {};
  stage(0, 0);
  for (int r = 0; r < ROWS; ++r){
    wait_async_lds();
    __syncthreads();                 // row r visible to all
    if (r + 1 < ROWS) stage((r+1)&1, r+1);   // overlap next-row staging
    const u16* buf = srow[r & 1];
    const int dz = r / KH, dy = r % KH;
    for (int dx = 0; dx < KW; ++dx){
      const int tap = (dz*KH + dy)*KW + dx;
      const u16* wbase = wp + ((size_t)tap*COUT + cout0 + mrow)*CINP;
      __builtin_prefetch(wbase, 0, 1);
      #pragma unroll
      for (int kc = 0; kc < KCH; ++kc){
        FragAB a0, a1, b;
        const u16* bp = wbase + kc*32;
        b.q[0] = *(const uint4*)(bp + kb);
        b.q[1] = *(const uint4*)(bp + kb + 16);
        const u16* ap0 = &buf[(mrow*SW + dx)*CINP + kc*32];
        a0.q[0] = *(const uint4*)(ap0 + kb);
        a0.q[1] = *(const uint4*)(ap0 + kb + 16);
        const u16* ap1 = &buf[((mrow+16)*SW + dx)*CINP + kc*32];
        a1.q[0] = *(const uint4*)(ap1 + kb);
        a1.q[1] = *(const uint4*)(ap1 + kb + 16);
        acc[0] = __builtin_amdgcn_wmma_f32_16x16x32_bf16(
                   false, a0.v, false, b.v, (short)0, acc[0], false, false);
        acc[1] = __builtin_amdgcn_wmma_f32_16x16x32_bf16(
                   false, a1.v, false, b.v, (short)0, acc[1], false, false);
      }
    }
    __syncthreads();                 // done reading buf before it is restaged
  }

  // fused epilogue: bias + folded BN + optional residual + ReLU + mask
  const int ch = cout0 + mrow;
  const float sc  = bng[ch] * rsqrtf(bnv[ch] + 1e-3f);
  const float sh_ = bnb[ch] - bnm[ch]*sc;
  const float bi  = bias ? bias[ch] : 0.f;
  #pragma unroll
  for (int half = 0; half < 2; ++half){
    #pragma unroll
    for (int r = 0; r < 8; ++r){
      const int mm = (lane >> 4)*8 + r;
      const int ox = x0 + half*16 + mm;
      if (ox >= Wout) continue;
      const size_t sp = ((size_t)oz*Hout + oy)*Wout + ox;
      float v = (acc[half][r] + bi)*sc + sh_;
      if (resid) v += bf2f(resid[sp*COUT + ch]);
      v = fmaxf(v, 0.f);
      if (mask) v *= bf2f(mask[sp]);
      out[sp*COUT + ch] = f2bf(v);
    }
  }
}

// ---------------- host orchestration ----------------
struct BNP { const float *g,*b,*m,*v; };
struct BlkP { const float *w1,*b1; BNP bn1; const float *w2,*b2; BNP bn2; };

template<int CIN,int COUT,int KD,int KH,int KW,int SD,int SH,int SW>
static void launch_conv(hipStream_t s, const u16* in, const u16* wp, u16* out,
                        const u16* mask, const u16* resid, const float* bias, const BNP& bn,
                        int Din,int Hin,int Win,int Dout,int Hout,int Wout,
                        int pd,int ph,int pw){
  dim3 g((Wout+31)/32, Hout, Dout), b(COUT*2);
  conv_wmma_k<CIN,COUT,KD,KH,KW,SD,SH,SW><<<g,b,0,s>>>(
    in,wp,out,mask,resid,bias,bn.g,bn.b,bn.m,bn.v,
    Din,Hin,Win,Dout,Hout,Wout,pd,ph,pw);
}

extern "C" void kernel_launch(void* const* d_in, const int* in_sizes, int n_in,
                              void* d_out, int out_size, void* d_ws, size_t ws_size,
                              hipStream_t stream){
  (void)n_in; (void)out_size; (void)ws_size;
  int ip = 0;
  const float* voxf = (const float*)d_in[ip++];
  const int*   coors = (const int*)d_in[ip++];
  auto F = [&](){ return (const float*)d_in[ip++]; };
  auto getBN  = [&](){ BNP p; p.g=F(); p.b=F(); p.m=F(); p.v=F(); return p; };
  auto getBlk = [&](){ BlkP k; k.w1=F(); k.b1=F(); k.bn1=getBN();
                               k.w2=F(); k.b2=F(); k.bn2=getBN(); return k; };
  const float* ciw = F(); BNP cibn = getBN();
  BlkP c1b0 = getBlk(), c1b1 = getBlk();
  const float* c2dw = F(); BNP c2dbn = getBN(); BlkP c2b0 = getBlk(), c2b1 = getBlk();
  const float* c3dw = F(); BNP c3dbn = getBN(); BlkP c3b0 = getBlk(), c3b1 = getBlk();
  const float* c4dw = F(); BNP c4dbn = getBN(); BlkP c4b0 = getBlk(), c4b1 = getBlk();
  const float* exw = F(); BNP exbn = getBN();

  const int N = in_sizes[1] / 4;

  // workspace bump allocator
  char* base = (char*)d_ws; size_t off = 0;
  auto alloc = [&](size_t bytes)->char*{ char* p = base + off;
                                         off += (bytes + 255) & ~(size_t)255; return p; };
  const size_t V0 = 41UL*200*200;
  float* acc0 = (float*)alloc(V0*16*4);
  float* cnt0 = (float*)alloc(V0*4);
  u16* x0 = (u16*)alloc(V0*16*2);
  u16* m0 = (u16*)alloc(V0*2);
  u16* m1 = (u16*)alloc(21UL*100*100*2);
  u16* m2 = (u16*)alloc(11UL*50*50*2);
  u16* m3 = (u16*)alloc(5UL*25*25*2);
  u16* m4 = (u16*)alloc(2UL*25*25*2);
  u16* A = (u16*)alloc(V0*16*2);
  u16* B = (u16*)alloc(V0*16*2);
  u16* C = (u16*)alloc(V0*16*2);
  u16* xex = (u16*)alloc(2UL*25*25*128*2);
  auto wal = [&](size_t el){ return (u16*)alloc(el*2); };
  u16* wci = wal(27*32*16);
  u16* w10 = wal(27*32*16),  *w11 = wal(27*32*16),  *w12 = wal(27*32*16),  *w13 = wal(27*32*16);
  u16* w2d = wal(27*32*32);
  u16* w20 = wal(27*32*32),  *w21 = wal(27*32*32),  *w22 = wal(27*32*32),  *w23 = wal(27*32*32);
  u16* w3d = wal(27*32*64);
  u16* w30 = wal(27*64*64),  *w31 = wal(27*64*64),  *w32 = wal(27*64*64),  *w33 = wal(27*64*64);
  u16* w4d = wal(27*64*128);
  u16* w40 = wal(27*128*128),*w41 = wal(27*128*128),*w42 = wal(27*128*128),*w43 = wal(27*128*128);
  u16* wex = wal(3*128*128);

  // scatter voxels -> dense bf16 + mask
  zero_f32<<<2048,256,0,stream>>>(acc0, V0*16);
  zero_f32<<<512,256,0,stream>>>(cnt0, V0);
  scatter_k<<<(N*16+255)/256,256,0,stream>>>(voxf, coors, acc0, cnt0, N);
  finalize_k<<<(int)((V0*16+255)/256),256,0,stream>>>(acc0, cnt0, x0, m0, V0);

  auto prep = [&](const float* w, u16* wpd, int CI,int CIP,int CO,int KD,int KH,int KW){
    int total = KD*KH*KW*CO*CIP;
    prep_w_k<<<(total+255)/256,256,0,stream>>>(w, wpd, CI, CIP, CO, KD, KH, KW);
  };
  prep(ciw,wci,16,32,16,3,3,3);
  prep(c1b0.w1,w10,16,32,16,3,3,3); prep(c1b0.w2,w11,16,32,16,3,3,3);
  prep(c1b1.w1,w12,16,32,16,3,3,3); prep(c1b1.w2,w13,16,32,16,3,3,3);
  prep(c2dw,w2d,16,32,32,3,3,3);
  prep(c2b0.w1,w20,32,32,32,3,3,3); prep(c2b0.w2,w21,32,32,32,3,3,3);
  prep(c2b1.w1,w22,32,32,32,3,3,3); prep(c2b1.w2,w23,32,32,32,3,3,3);
  prep(c3dw,w3d,32,32,64,3,3,3);
  prep(c3b0.w1,w30,64,64,64,3,3,3); prep(c3b0.w2,w31,64,64,64,3,3,3);
  prep(c3b1.w1,w32,64,64,64,3,3,3); prep(c3b1.w2,w33,64,64,64,3,3,3);
  prep(c4dw,w4d,64,64,128,3,3,3);
  prep(c4b0.w1,w40,128,128,128,3,3,3); prep(c4b0.w2,w41,128,128,128,3,3,3);
  prep(c4b1.w1,w42,128,128,128,3,3,3); prep(c4b1.w2,w43,128,128,128,3,3,3);
  prep(exw,wex,128,128,128,3,1,1);

  // mask pyramid
  down_mask_k<<<(21*100*100+255)/256,256,0,stream>>>(m0,m1,41,200,200,21,100,100,3,3,3,2,2,2,1,1,1);
  down_mask_k<<<(11*50*50+255)/256,256,0,stream>>>(m1,m2,21,100,100,11,50,50,3,3,3,2,2,2,1,1,1);
  down_mask_k<<<(5*25*25+255)/256,256,0,stream>>>(m2,m3,11,50,50,5,25,25,3,3,3,2,2,2,0,1,1);
  down_mask_k<<<(2*25*25+255)/256,256,0,stream>>>(m3,m4,5,25,25,2,25,25,3,1,1,2,1,1,0,0,0);

  // stage 1 (16ch, 41x200x200)
  launch_conv<16,16,3,3,3,1,1,1>(stream, x0, wci, A, m0, nullptr, nullptr, cibn, 41,200,200,41,200,200,1,1,1);
  launch_conv<16,16,3,3,3,1,1,1>(stream, A, w10, B, m0, nullptr, c1b0.b1, c1b0.bn1, 41,200,200,41,200,200,1,1,1);
  launch_conv<16,16,3,3,3,1,1,1>(stream, B, w11, C, m0, A,       c1b0.b2, c1b0.bn2, 41,200,200,41,200,200,1,1,1);
  launch_conv<16,16,3,3,3,1,1,1>(stream, C, w12, B, m0, nullptr, c1b1.b1, c1b1.bn1, 41,200,200,41,200,200,1,1,1);
  launch_conv<16,16,3,3,3,1,1,1>(stream, B, w13, A, m0, C,       c1b1.b2, c1b1.bn2, 41,200,200,41,200,200,1,1,1);
  // stage 2 (32ch, 21x100x100)
  launch_conv<16,32,3,3,3,2,2,2>(stream, A, w2d, B, m1, nullptr, nullptr, c2dbn, 41,200,200,21,100,100,1,1,1);
  launch_conv<32,32,3,3,3,1,1,1>(stream, B, w20, C, m1, nullptr, c2b0.b1, c2b0.bn1, 21,100,100,21,100,100,1,1,1);
  launch_conv<32,32,3,3,3,1,1,1>(stream, C, w21, A, m1, B,       c2b0.b2, c2b0.bn2, 21,100,100,21,100,100,1,1,1);
  launch_conv<32,32,3,3,3,1,1,1>(stream, A, w22, C, m1, nullptr, c2b1.b1, c2b1.bn1, 21,100,100,21,100,100,1,1,1);
  launch_conv<32,32,3,3,3,1,1,1>(stream, C, w23, B, m1, A,       c2b1.b2, c2b1.bn2, 21,100,100,21,100,100,1,1,1);
  // stage 3 (64ch, 11x50x50)
  launch_conv<32,64,3,3,3,2,2,2>(stream, B, w3d, A, m2, nullptr, nullptr, c3dbn, 21,100,100,11,50,50,1,1,1);
  launch_conv<64,64,3,3,3,1,1,1>(stream, A, w30, C, m2, nullptr, c3b0.b1, c3b0.bn1, 11,50,50,11,50,50,1,1,1);
  launch_conv<64,64,3,3,3,1,1,1>(stream, C, w31, B, m2, A,       c3b0.b2, c3b0.bn2, 11,50,50,11,50,50,1,1,1);
  launch_conv<64,64,3,3,3,1,1,1>(stream, B, w32, C, m2, nullptr, c3b1.b1, c3b1.bn1, 11,50,50,11,50,50,1,1,1);
  launch_conv<64,64,3,3,3,1,1,1>(stream, C, w33, A, m2, B,       c3b1.b2, c3b1.bn2, 11,50,50,11,50,50,1,1,1);
  // stage 4 (128ch, 5x25x25), pad (0,1,1) on downsample
  launch_conv<64,128,3,3,3,2,2,2>(stream, A, w4d, B, m3, nullptr, nullptr, c4dbn, 11,50,50,5,25,25,0,1,1);
  launch_conv<128,128,3,3,3,1,1,1>(stream, B, w40, C, m3, nullptr, c4b0.b1, c4b0.bn1, 5,25,25,5,25,25,1,1,1);
  launch_conv<128,128,3,3,3,1,1,1>(stream, C, w41, A, m3, B,       c4b0.b2, c4b0.bn2, 5,25,25,5,25,25,1,1,1);
  launch_conv<128,128,3,3,3,1,1,1>(stream, A, w42, C, m3, nullptr, c4b1.b1, c4b1.bn1, 5,25,25,5,25,25,1,1,1);
  launch_conv<128,128,3,3,3,1,1,1>(stream, C, w43, B, m3, A,       c4b1.b2, c4b1.bn2, 5,25,25,5,25,25,1,1,1);
  // extra (3,1,1) stride (2,1,1) pad 0 -> (2,25,25,128)
  launch_conv<128,128,3,1,1,2,1,1>(stream, B, wex, xex, m4, nullptr, nullptr, exbn, 5,25,25,2,25,25,0,0,0);
  // reshape to (256,25,25) fp32
  final_out_k<<<(256*25*25+255)/256,256,0,stream>>>(xex, (float*)d_out);
}
